// LSTMModel_34497177321571
// MI455X (gfx1250) — matrix operand order repaired and checked by reading, compiled, not run on hardware
//
#include <hip/hip_runtime.h>

// ---------------------------------------------------------------------------
// Persistent-LSTM decoder for MI455X (gfx1250, wave32, WMMA).
// Weights live in LDS (bf16) across the whole 200-step recurrence, staged by
// the Tensor Data Mover; activations bf16; f32 accumulation via
// v_wmma_f32_16x16x32_bf16.
// ---------------------------------------------------------------------------

#define USE_TDM 1
#define DEVINL __device__ __forceinline__

typedef __attribute__((ext_vector_type(16))) __bf16        bf16x16;
typedef __attribute__((ext_vector_type(8)))  float         f32x8;
typedef __attribute__((ext_vector_type(4)))  unsigned int  u32x4;
typedef __attribute__((ext_vector_type(8)))  int           i32x8;
typedef __attribute__((ext_vector_type(4)))  int           i32x4;

union Frag16 { bf16x16 v; u32x4 q[2]; };

DEVINL f32x8 wmma_bf16(bf16x16 a, bf16x16 b, f32x8 c) {
  // 8 args: (neg_a, A, neg_b, B, c_mod, C, reuse_a, reuse_b)
  return __builtin_amdgcn_wmma_f32_16x16x32_bf16(false, a, false, b,
                                                 (short)0, c, false, false);
}

DEVINL unsigned short f2bf(float f) {  // round-to-nearest-even f32 -> bf16
  unsigned u = __float_as_uint(f);
  u += 0x7FFFu + ((u >> 16) & 1u);
  return (unsigned short)(u >> 16);
}
DEVINL float sigm(float x) { return 1.0f / (1.0f + __expf(-x)); }

// ----- problem dimensions ---------------------------------------------------
constexpr int B   = 256;
constexpr int T   = 201;
constexpr int H   = 512;
constexpr int L   = 3;
constexpr int V   = 100;
constexpr int G4  = 4 * H;      // 2048 gate columns per layer
constexpr int TS  = T - 1;      // 200 recurrent steps
constexpr int SOS = 1;

// ----- persistent-kernel partitioning --------------------------------------
constexpr int NWG = 128;        // persistent workgroups (grid barrier members)
constexpr int HS  = H / NWG;    // 4 hidden units per WG
constexpr int GS  = 4 * HS;     // 16 gate columns per WG (i,f,g,o blocks of HS)
constexpr int RESR = 208;       // 201 rows padded to 13 M-tiles of 16
constexpr int VP   = 112;       // 100 vocab padded to 7 N-tiles of 16

// ----- LDS layout (dynamic) -------------------------------------------------
constexpr int LDS_WROWS  = L * 2 * GS;            // 96 weight rows (bf16, K=512)
constexpr int LDS_WBYTES = LDS_WROWS * H * 2;     // 98304 B
constexpr int LDS_GBYTES = B * GS * 4;            // 16384 B gate scratch (f32)
constexpr int LDS_TOTAL  = LDS_WBYTES + LDS_GBYTES; // 114688 B (< 320KB WGP LDS)

// ----- workspace byte offsets ----------------------------------------------
constexpr size_t SZ_WB    = (size_t)L * G4 * H * 2;           // one bf16 weight set
constexpr size_t OFF_WIH  = 0;
constexpr size_t OFF_WHH  = OFF_WIH + SZ_WB;
constexpr size_t OFF_BIAS = OFF_WHH + SZ_WB;                  // L*G4 f32
constexpr size_t OFF_EMB  = OFF_BIAS + (size_t)L * G4 * 4;    // V*H bf16
constexpr size_t OFF_PW   = OFF_EMB + (size_t)V * H * 2;      // VP*H bf16 (zero padded)
constexpr size_t OFF_X    = OFF_PW + (size_t)VP * H * 2;      // TS*B*H bf16
constexpr size_t OFF_H    = OFF_X + (size_t)TS * B * H * 2;   // 2*L*B*H bf16 (ping-pong)
constexpr size_t OFF_C    = OFF_H + (size_t)2 * L * B * H * 2;// L*B*H f32
constexpr size_t OFF_RES  = OFF_C + (size_t)L * B * H * 4;    // B*RESR*H bf16
constexpr size_t OFF_SYNC = OFF_RES + (size_t)B * RESR * H * 2; // barrier state

// ---------------------------------------------------------------------------
// Grid-wide sense-reversing barrier (all NWG workgroups resident).
// ---------------------------------------------------------------------------
DEVINL void grid_sync(unsigned* cnt, unsigned* gen, unsigned nwg) {
  __syncthreads();
  if (threadIdx.x == 0) {
    __threadfence();
    unsigned g = __hip_atomic_load(gen, __ATOMIC_ACQUIRE, __HIP_MEMORY_SCOPE_AGENT);
    unsigned a = __hip_atomic_fetch_add(cnt, 1u, __ATOMIC_ACQ_REL, __HIP_MEMORY_SCOPE_AGENT);
    if (a == nwg - 1u) {
      __hip_atomic_store(cnt, 0u, __ATOMIC_RELAXED, __HIP_MEMORY_SCOPE_AGENT);
      __hip_atomic_store(gen, g + 1u, __ATOMIC_RELEASE, __HIP_MEMORY_SCOPE_AGENT);
    } else {
      while (__hip_atomic_load(gen, __ATOMIC_ACQUIRE, __HIP_MEMORY_SCOPE_AGENT) == g)
        __builtin_amdgcn_s_sleep(2);
    }
  }
  __syncthreads();
}

#if USE_TDM
// ---------------------------------------------------------------------------
// Issue one TDM 3D tile load: x = 512 contiguous bf16 (one K row),
// y = HS(4) rows at stride H, z = 4 gate groups (i,f,g,o) at stride 512*H.
// Lands contiguously at LDS byte offset `lds_off`. D# layout per ISA ch.8.
// This toolchain exposes the 6-arg builtin: (g0, g1, g2, g3, g4, cpol).
// ---------------------------------------------------------------------------
DEVINL void tdm_load_wslice(const unsigned short* gsrc, unsigned lds_off) {
  const unsigned long long ga = (unsigned long long)(const void*)gsrc;
  u32x4 g0;
  g0[0] = 1u;                                        // count=1, user load D#
  g0[1] = lds_off;                                   // lds_addr (bytes)
  g0[2] = (unsigned)(ga & 0xFFFFFFFFu);              // global_addr[31:0]
  g0[3] = (unsigned)((ga >> 32) & 0x1FFFFFFu)        // global_addr[56:32]
        | (2u << 30);                                // type = 2 (image)
  i32x8 g1;
  g1[0] = (int)(1u << 16);                           // wg_mask=0, data_size=2B
  g1[1] = (int)((unsigned)H << 16);                  // tensor_dim0 = 512 (lo16)
  g1[2] = (int)(((unsigned)G4 & 0xFFFFu) << 16);     // td0 hi=0 | tensor_dim1=2048 lo
  g1[3] = (int)((unsigned)H << 16);                  // td1 hi=0 | tile_dim0 = 512
  g1[4] = (int)((unsigned)HS | (4u << 16));          // tile_dim1=4 | tile_dim2=4
  g1[5] = (int)(unsigned)H;                          // tensor_dim0_stride = 512
  g1[6] = (int)(((unsigned)(512 * H) & 0xFFFFu) << 16); // s0 hi=0 | s1[15:0]
  g1[7] = (int)((unsigned)(512 * H) >> 16);          // tensor_dim1_stride[47:16]
  i32x4 g2;
  g2[0] = 4;                                         // tensor_dim2 = 4
  g2[1] = 0; g2[2] = 0; g2[3] = 0;                   // no dim3 / no iterate
  i32x4 g3; g3[0] = 0; g3[1] = 0; g3[2] = 0; g3[3] = 0;
  i32x8 g4;                                          // unused trailing group
  g4[0] = 0; g4[1] = 0; g4[2] = 0; g4[3] = 0;
  g4[4] = 0; g4[5] = 0; g4[6] = 0; g4[7] = 0;
  __builtin_amdgcn_tensor_load_to_lds(g0, g1, g2, g3, g4, 0);
}
#endif

// ---------------------------------------------------------------------------
// Prep 1: f32 -> bf16 weight/embedding/projection conversion, bias fusion.
// ---------------------------------------------------------------------------
__global__ void k_prep1(const float* __restrict__ Wih, const float* __restrict__ Whh,
                        const float* __restrict__ bih, const float* __restrict__ bhh,
                        const float* __restrict__ emb, const float* __restrict__ pw,
                        unsigned short* __restrict__ WihB, unsigned short* __restrict__ WhhB,
                        float* __restrict__ biasS, unsigned short* __restrict__ embB,
                        unsigned short* __restrict__ pwB) {
  const size_t g  = (size_t)blockIdx.x * blockDim.x + threadIdx.x;
  const size_t gs = (size_t)gridDim.x * blockDim.x;
  const size_t NW = (size_t)L * G4 * H;
  for (size_t i = g; i < NW; i += gs) { WihB[i] = f2bf(Wih[i]); WhhB[i] = f2bf(Whh[i]); }
  for (size_t i = g; i < (size_t)L * G4; i += gs) biasS[i] = bih[i] + bhh[i];
  for (size_t i = g; i < (size_t)V * H; i += gs) embB[i] = f2bf(emb[i]);
  for (size_t i = g; i < (size_t)VP * H; i += gs) {
    size_t v = i >> 9;                       // H = 512
    pwB[i] = (v < (size_t)V) ? f2bf(pw[i]) : (unsigned short)0;
  }
}

// ---------------------------------------------------------------------------
// Prep 2: embedding gather for all steps, h0/c0 init, res row 0, barrier reset.
// ---------------------------------------------------------------------------
__global__ void k_prep2(const float* __restrict__ feat, const int* __restrict__ tseq,
                        const unsigned short* __restrict__ embB,
                        unsigned short* __restrict__ X, unsigned short* __restrict__ hbuf,
                        float* __restrict__ cbuf, unsigned short* __restrict__ res,
                        unsigned* __restrict__ sync) {
  const size_t g  = (size_t)blockIdx.x * blockDim.x + threadIdx.x;
  const size_t gs = (size_t)gridDim.x * blockDim.x;
  // X[t][b][:] = embed[target_seq[b][t+1]] as bf16, 16B vectors
  const size_t NX = (size_t)TS * B * (H / 8);
  for (size_t i = g; i < NX; i += gs) {
    size_t row = i >> 6;                     // H/8 = 64 chunks per row
    int    c8  = (int)(i & 63);
    int    t   = (int)(row / B), b = (int)(row % B);
    int    tok = tseq[(size_t)b * T + t + 1];
    ((u32x4*)X)[row * 64 + c8] = ((const u32x4*)embB)[(size_t)tok * 64 + c8];
  }
  // h0 (ping buffer 0) = feat for every layer; c0 = 0
  for (size_t i = g; i < (size_t)L * B * H; i += gs) {
    hbuf[i] = f2bf(feat[i % ((size_t)B * H)]);
    cbuf[i] = 0.0f;
  }
  // res[:, 0, :] = embed[SOS]
  for (size_t i = g; i < (size_t)B * H; i += gs) {
    size_t b = i >> 9; int k = (int)(i & 511);
    res[(b * RESR) * H + k] = embB[(size_t)SOS * H + k];
  }
  if (g == 0) { sync[0] = 0u; sync[1] = 0u; }
}

// ---------------------------------------------------------------------------
// Persistent LSTM: 128 WGs x 256 threads (8 waves). Each WG owns hidden units
// [wg*HS, wg*HS+HS) of every layer; its W_ih/W_hh gate-row slices (i,f,g,o)
// are DMA'd into LDS once by the TDM. Per (t, layer): 8 waves x 64 WMMA.
// ---------------------------------------------------------------------------
__global__ __launch_bounds__(256, 1) void k_lstm(
    const unsigned short* __restrict__ WihB, const unsigned short* __restrict__ WhhB,
    const float* __restrict__ bias, const unsigned short* __restrict__ X,
    unsigned short* __restrict__ hbuf, float* __restrict__ cbuf,
    unsigned short* __restrict__ res, unsigned* __restrict__ sync) {
  extern __shared__ char smem[];
  float* gl = (float*)(smem + LDS_WBYTES);

  const int tid  = threadIdx.x;
  const int wg   = blockIdx.x;
  const int wid  = tid >> 5;
  const int lane = tid & 31;
  const int l15  = lane & 15;
  const int kA   = (lane & 16) ? 8 : 0;     // A-frag K base (and C-row half)
  const int kB   = (lane & 16) ? 16 : 0;    // unused; B-frag uses kA as well

#if USE_TDM
  // ---- stage weight slices into LDS via Tensor Data Mover (one DMA per
  //      (layer, matrix) pair, issued by waves 0..5; EXEC is ignored by TDM).
  if (wid < L * 2) {
    const int ll = wid >> 1, m = wid & 1;
    const unsigned short* W = m ? WhhB : WihB;
    const unsigned short* gsrc = W + ((size_t)ll * G4 + (size_t)wg * HS) * H;
    tdm_load_wslice(gsrc, (unsigned)((ll * 2 + m) * GS * (H * 2)));
  }
  __builtin_amdgcn_s_wait_tensorcnt(0);
#else
  for (int i = tid; i < LDS_WROWS * (H / 8); i += 256) {
    int row = i >> 6, c8 = i & 63;
    int ll = row / (2 * GS), rem = row % (2 * GS);
    int m = rem / GS, r = rem % GS;
    int q = r / HS, j = r % HS;
    const unsigned short* W = m ? WhhB : WihB;
    size_t grow = (size_t)ll * G4 + (size_t)q * H + (size_t)wg * HS + j;
    *(u32x4*)(smem + (size_t)row * (H * 2) + (size_t)c8 * 16) =
        *(const u32x4*)(W + grow * H + (size_t)c8 * 8);
  }
#endif
  __syncthreads();

  int pOld = 0;
  for (int t = 0; t < TS; ++t) {
    const int pNew = pOld ^ 1;
    if (t + 1 < TS) {  // prefetch next step's embedded inputs (global_prefetch)
      const unsigned short* nx = X + (size_t)(t + 1) * B * H;
      #pragma unroll
      for (int i = 0; i < 8; ++i)
        __builtin_prefetch(nx + ((size_t)tid * 8 + i) * 64, 0, 0);
    }
    for (int l = 0; l < L; ++l) {
      const unsigned short* inp = (l == 0)
          ? (X + (size_t)t * B * H)
          : (hbuf + (size_t)(pNew * L + (l - 1)) * B * H);
      const unsigned short* hpr = hbuf + (size_t)(pOld * L + l) * B * H;

      f32x8 acc0 = {0, 0, 0, 0, 0, 0, 0, 0};
      f32x8 acc1 = {0, 0, 0, 0, 0, 0, 0, 0};
      for (int s = 0; s < 2; ++s) {          // s=0: x @ Wih^T, s=1: h @ Whh^T
        const unsigned short* Ap = s ? hpr : inp;
        const char* Wb = smem + (size_t)(l * 2 + s) * GS * (H * 2);
        #pragma unroll 4
        for (int kb = 0; kb < H / 32; ++kb) {
          Frag16 bfr;                        // B: 32x16 slice of W^T from LDS
          const char* bp = Wb + (size_t)l15 * (H * 2) + kb * 64 + kA * 2;
          bfr.q[0] = *(const u32x4*)bp;
          bfr.q[1] = *(const u32x4*)(bp + 32);
          {
            const unsigned short* ap =
                Ap + (size_t)((wid * 2 + 0) * 16 + l15) * H + kb * 32 + kA;
            Frag16 af; af.q[0] = *(const u32x4*)ap; af.q[1] = *(const u32x4*)(ap + 16);
            acc0 = wmma_bf16(af.v, bfr.v, acc0);
          }
          {
            const unsigned short* ap =
                Ap + (size_t)((wid * 2 + 1) * 16 + l15) * H + kb * 32 + kA;
            Frag16 af; af.q[0] = *(const u32x4*)ap; af.q[1] = *(const u32x4*)(ap + 16);
            acc1 = wmma_bf16(af.v, bfr.v, acc1);
          }
        }
      }
      // gates -> LDS with fused (b_ih + b_hh)
      {
        const int col = l15;
        const float bz = bias[(size_t)l * G4 + (size_t)(col / HS) * H + wg * HS + (col % HS)];
        const int rb0 = (wid * 2 + 0) * 16 + kA;
        const int rb1 = (wid * 2 + 1) * 16 + kA;
        #pragma unroll
        for (int r = 0; r < 8; ++r) {
          gl[(rb0 + r) * GS + col] = acc0[r] + bz;
          gl[(rb1 + r) * GS + col] = acc1[r] + bz;
        }
      }
      __syncthreads();
      // pointwise LSTM cell for this WG's hidden-unit slice
      for (int it = tid; it < B * HS; it += 256) {
        int b = it / HS, hu = it % HS;
        const float* g = gl + b * GS;
        float gi = sigm(g[0 * HS + hu]);
        float gf = sigm(g[1 * HS + hu]);
        float gg = tanhf(g[2 * HS + hu]);
        float go = sigm(g[3 * HS + hu]);
        size_t cix = ((size_t)l * B + b) * H + (size_t)wg * HS + hu;
        float cn = gf * cbuf[cix] + gi * gg;
        float hn = go * tanhf(cn);
        cbuf[cix] = cn;
        unsigned short hb = f2bf(hn);
        hbuf[((size_t)(pNew * L + l) * B + b) * H + (size_t)wg * HS + hu] = hb;
        if (l == L - 1)
          res[((size_t)b * RESR + (t + 1)) * H + (size_t)wg * HS + hu] = hb;
      }
      grid_sync(sync, sync + 1, NWG);
    }
    pOld = pNew;
  }
}

// ---------------------------------------------------------------------------
// Projection: per batch element, (RESR x 512) @ (512 x VP) with WMMA, bias,
// transposed store to out[b][v][t].
// ---------------------------------------------------------------------------
__global__ __launch_bounds__(256) void k_proj(const unsigned short* __restrict__ res,
                                              const unsigned short* __restrict__ pwB,
                                              const float* __restrict__ pb,
                                              float* __restrict__ out) {
  const int b    = blockIdx.x;
  const int tid  = threadIdx.x;
  const int wid  = tid >> 5;
  const int lane = tid & 31;
  const int l15  = lane & 15;
  const int kA   = (lane & 16) ? 8 : 0;
  const int kB   = (lane & 16) ? 16 : 0;
  const unsigned short* A = res + (size_t)b * RESR * H;

  for (int tile = wid; tile < (RESR / 16) * (VP / 16); tile += 8) {
    int mt = tile / (VP / 16), nt = tile % (VP / 16);
    f32x8 acc = {0, 0, 0, 0, 0, 0, 0, 0};
    #pragma unroll 4
    for (int kb = 0; kb < H / 32; ++kb) {
      Frag16 bfr;
      const unsigned short* bp =
          pwB + (size_t)(nt * 16 + l15) * H + kb * 32 + kA;
      bfr.q[0] = *(const u32x4*)bp;
      bfr.q[1] = *(const u32x4*)(bp + 16);
      Frag16 af;
      const unsigned short* ap =
          A + (size_t)(mt * 16 + l15) * H + kb * 32 + kA;
      af.q[0] = *(const u32x4*)ap;
      af.q[1] = *(const u32x4*)(ap + 16);
      acc = wmma_bf16(af.v, bfr.v, acc);
    }
    int v = nt * 16 + l15;
    if (v < V) {
      float bz = pb[v];
      int tbase = mt * 16 + kA;
      #pragma unroll
      for (int r = 0; r < 8; ++r) {
        int tt = tbase + r;
        if (tt < T) out[((size_t)b * V + v) * T + tt] = acc[r] + bz;
      }
    }
  }
}

// ---------------------------------------------------------------------------
extern "C" void kernel_launch(void* const* d_in, const int* in_sizes, int n_in,
                              void* d_out, int out_size, void* d_ws, size_t ws_size,
                              hipStream_t stream) {
  (void)in_sizes; (void)n_in; (void)out_size; (void)ws_size;
  const float* feat = (const float*)d_in[0];
  const int*   tseq = (const int*)d_in[1];
  const float* emb  = (const float*)d_in[2];
  const float* Wih  = (const float*)d_in[3];
  const float* Whh  = (const float*)d_in[4];
  const float* bih  = (const float*)d_in[5];
  const float* bhh  = (const float*)d_in[6];
  const float* pw   = (const float*)d_in[7];
  const float* pb   = (const float*)d_in[8];
  float* out = (float*)d_out;
  char*  ws  = (char*)d_ws;

  unsigned short* WihB = (unsigned short*)(ws + OFF_WIH);
  unsigned short* WhhB = (unsigned short*)(ws + OFF_WHH);
  float*          bS   = (float*)(ws + OFF_BIAS);
  unsigned short* embB = (unsigned short*)(ws + OFF_EMB);
  unsigned short* pwB  = (unsigned short*)(ws + OFF_PW);
  unsigned short* X    = (unsigned short*)(ws + OFF_X);
  unsigned short* hbuf = (unsigned short*)(ws + OFF_H);
  float*          cbuf = (float*)(ws + OFF_C);
  unsigned short* resB = (unsigned short*)(ws + OFF_RES);
  unsigned*       sync = (unsigned*)(ws + OFF_SYNC);

  k_prep1<<<1024, 256, 0, stream>>>(Wih, Whh, bih, bhh, emb, pw,
                                    WihB, WhhB, bS, embB, pwB);
  k_prep2<<<2048, 256, 0, stream>>>(feat, tseq, embB, X, hbuf, cbuf, resB, sync);
  k_lstm<<<NWG, 256, LDS_TOTAL, stream>>>(WihB, WhhB, bS, X, hbuf, cbuf, resB, sync);
  k_proj<<<B, 256, 0, stream>>>(resB, pwB, pb, out);
}
